// CalculateVector_62801011802517
// MI455X (gfx1250) — matrix operand order repaired and checked by reading, not compiled, not run on hardware
//
#include <hip/hip_runtime.h>

typedef _Float16 v16h __attribute__((ext_vector_type(16)));
typedef _Float16 v8h  __attribute__((ext_vector_type(8)));
typedef _Float16 v4h  __attribute__((ext_vector_type(4)));
typedef float    v8f  __attribute__((ext_vector_type(8)));
typedef short    v8s  __attribute__((ext_vector_type(8)));

#define B_   2
#define H_   270
#define W_   480
#define N_   26
#define K2_  16
#define NP   (B_*H_*W_)          // 259200 pixels
#define NG   (NP/16)             // 16200 groups of 16 pixels (one wave each)
#define ROWB (N_*K2_*2)          // 832 bytes per pixel (26 rows x 16 f16)

// d_out float sections (concatenated tuple, converted to f32):
#define OUT_VEC  0               // (NP,2)  vector
#define OUT_TMPL (2*NP)          // (NP,16) min_templates
#define OUT_MASK (18*NP)         // (NP,)   input_mv_mask (0/1)
#define OUT_COST (19*NP)         // (NP,)   min_cost_volume

// Spiral ranks for flat idx 0..24, packed 8b LSB-first:
//   RANK = [20,21,22,23,24, 19,6,7,8,9, 18,5,0,1,10, 17,4,3,2,11, 16,15,14,13,12]
#define PACK_A 0x0706131817161514ULL   // idx 0..7
#define PACK_B 0x110A010005120908ULL   // idx 8..15
#define PACK_C 0x0D0E0F100B020304ULL   // idx 16..23 (idx24 -> 12)

__global__ __launch_bounds__(256)
void bm_sad_wmma_kernel(const _Float16* __restrict__ w1,
                        const _Float16* __restrict__ w2,
                        float* __restrict__ out) {
    const int wave = threadIdx.x >> 5;
    const int lane = threadIdx.x & 31;
    const int g    = blockIdx.x * 8 + wave;       // pixel-group id (uniform per wave)
    if (g >= NG) return;

    const int sub   = lane & 15;                  // A-matrix row = pixel in group
    const int kbase = (lane >> 4) * 8;            // low lanes k=0..7, high lanes k=8..15
    const char* w1b = (const char*)w1;
    const char* w2b = (const char*)w2;
    const long  pixBase = (long)(g * 16 + sub) * ROWB;

    const v8s abs_mask = {0x7FFF,0x7FFF,0x7FFF,0x7FFF,0x7FFF,0x7FFF,0x7FFF,0x7FFF};

    v8f d0 = {};   // costs for candidates 0..15  (column = lane&15)
    v8f d1 = {};   // costs for candidates 16..25 (column = lane&15, cols 10..15 unused)

    #pragma unroll
    for (int t = 0; t < 13; ++t) {
        const int  n0 = 2*t, n1 = 2*t + 1;        // two candidate rows per WMMA
        const long o0 = pixBase + (long)n0 * 32 + kbase * 2;
        const long o1 = pixBase + (long)n1 * 32 + kbase * 2;
        v8h a0 = *(const v8h*)(w1b + o0);
        v8h b0 = *(const v8h*)(w2b + o0);
        v8h a1 = *(const v8h*)(w1b + o1);
        v8h b1 = *(const v8h*)(w2b + o1);
        v8h e0 = a0 - b0;                          // exact: small ints in f16
        v8h e1 = a1 - b1;
        v8s s0 = __builtin_bit_cast(v8s, e0) & abs_mask;   // |x| = clear sign bits
        v8s s1 = __builtin_bit_cast(v8s, e1) & abs_mask;
        e0 = __builtin_bit_cast(v8h, s0);
        e1 = __builtin_bit_cast(v8h, s1);
        // A frag: halves 0..7 -> K (n0 side), halves 8..15 -> K+16 (n1 side)
        v16h af = __builtin_shufflevector(e0, e1, 0,1,2,3,4,5,6,7,8,9,10,11,12,13,14,15);

        // B selector: ones in column c0 for K<16 (elements 0..7) and column c0+1 for K>=16 (elements 8..15), both lane halves
        const int c0 = (t < 8) ? 2*t : 2*(t - 8);
        v16h bf;
        #pragma unroll
        for (int e = 0; e < 16; ++e) bf[e] = (sub == ((e < 8) ? c0 : c0 + 1))
                                             ? (_Float16)1.0f : (_Float16)0.0f;

        if (t < 8)
            d0 = __builtin_amdgcn_wmma_f32_16x16x32_f16(false, af, false, bf,
                                                        (short)0, d0, false, false);
        else
            d1 = __builtin_amdgcn_wmma_f32_16x16x32_f16(false, af, false, bf,
                                                        (short)0, d1, false, false);
    }

    // Per-lane spiral ranks (lane&15 = candidate column)
    const int rank_lo = (int)(((sub < 8) ? (PACK_A >> (8*sub))
                                         : (PACK_B >> (8*(sub-8)))) & 0xFF);
    const int rank_hi = (sub < 8) ? (int)((PACK_C >> (8*sub)) & 0xFF) : 12; // idx 16+sub
    const int hbase   = (lane & 16) ? 8 : 0;      // D rows: low half m=r, high half m=r+8

    #pragma unroll
    for (int r = 0; r < 8; ++r) {
        // Build tie-break key: (cost<<16) | (spiral_rank<<8) | idx  (min => argmin w/ spiral prio)
        const int cA  = (int)(d0[r] + 0.5f);
        int key = (cA << 16) | (rank_lo << 8) | sub;
        const int cB  = (int)(d1[r] + 0.5f);
        const int keyB = (sub <= 8) ? ((cB << 16) | (rank_hi << 8) | (16 + sub))
                                    : 0x7FFFFFFF;            // cols 9..15: n=25 (MV) / unused
        if (keyB < key) key = keyB;

        // min-reduce within each 16-lane half (one pixel per half)
        #pragma unroll
        for (int m = 1; m < 16; m <<= 1) {
            const int o = __shfl_xor(key, m, 32);
            if (o < key) key = o;
        }

        // input-MV cost lives in D1 column 9 of this half
        const float ciF = __shfl(d1[r], (lane & 16) | 9, 32);
        const int   ci  = (int)(ciF + 0.5f);

        const int idx_bm   = key & 0xFF;
        const int mcb      = key >> 16;
        const bool m_in    = ci < mcb;
        const int  min_idx = m_in ? 25 : idx_bm;
        const int  min_cst = m_in ? ci : mcb;
        const int  p       = g * 16 + hbase + r;

        if (sub == 0) {
            out[OUT_VEC + 2*p + 0] = (float)(2 - idx_bm / 5);  // -jj
            out[OUT_VEC + 2*p + 1] = (float)(2 - idx_bm % 5);  // -ii
            out[OUT_MASK + p] = m_in ? 1.0f : 0.0f;
            out[OUT_COST + p] = (float)min_cst;
        }
        if (sub < 4) {  // 4 lanes x 4 f16 -> 16-float winning template row
            const v4h t4 = *(const v4h*)(w1b + (long)p * ROWB + min_idx * 32 + sub * 8);
            float4 f4 = make_float4((float)t4[0], (float)t4[1], (float)t4[2], (float)t4[3]);
            *(float4*)(out + OUT_TMPL + p * 16 + sub * 4) = f4;
        }
    }
}

extern "C" void kernel_launch(void* const* d_in, const int* in_sizes, int n_in,
                              void* d_out, int out_size, void* d_ws, size_t ws_size,
                              hipStream_t stream) {
    (void)in_sizes; (void)n_in; (void)out_size; (void)d_ws; (void)ws_size;
    const _Float16* w1 = (const _Float16*)d_in[0];
    const _Float16* w2 = (const _Float16*)d_in[1];
    float* out = (float*)d_out;
    dim3 block(256);                  // 8 waves/block, one 16-pixel group per wave
    dim3 grid((NG + 7) / 8);          // 2025 blocks
    bm_sad_wmma_kernel<<<grid, block, 0, stream>>>(w1, w2, out);
}
